// LocalAttention_16690242912348
// MI455X (gfx1250) — matrix operand order repaired and checked by reading, not compiled, not run on hardware
//
#include <hip/hip_runtime.h>
#include <hip/hip_fp16.h>
#include <hip/hip_bf16.h>

typedef __attribute__((ext_vector_type(16))) _Float16 v16h;
typedef __attribute__((ext_vector_type(8)))  float    v8f;

#define H    768
#define NH   12
#define HD   64
#define WS   128
#define SEQ  4096
#define BS   4
#define MTOT (SEQ * BS)   // 16384 rows in the projection GEMMs

// low 32 bits of a generic pointer to LDS == wave-relative LDS byte address
__device__ __forceinline__ unsigned lds_addr32(const void* p) {
    return (unsigned)(unsigned long long)(uintptr_t)p;
}

// ---------------------------------------------------------------------------
// Kernel 0a: hs fp32 -> f16
// ---------------------------------------------------------------------------
__global__ __launch_bounds__(256) void convert_hs(
    const float* __restrict__ hs, _Float16* __restrict__ hsf)
{
    for (size_t i = (size_t)blockIdx.x * 256 + threadIdx.x;
         i < (size_t)MTOT * H; i += (size_t)gridDim.x * 256)
        hsf[i] = (_Float16)hs[i];
}

// ---------------------------------------------------------------------------
// Kernel 0b: W fp32 -> f16, transposed: Wt[z][n][k] = W_z[k][n]
// ---------------------------------------------------------------------------
__global__ __launch_bounds__(256) void convert_w(
    const float* __restrict__ Wq, const float* __restrict__ Wk,
    const float* __restrict__ Wv, _Float16* __restrict__ Wt)
{
    for (size_t i = (size_t)blockIdx.x * 256 + threadIdx.x;
         i < (size_t)3 * H * H; i += (size_t)gridDim.x * 256) {
        int z   = (int)(i / (H * H));
        int rem = (int)(i % (H * H));
        int n   = rem / H;
        int k   = rem % H;
        const float* W = (z == 0) ? Wq : (z == 1) ? Wk : Wv;
        Wt[i] = (_Float16)W[(size_t)k * H + n];
    }
}

// ---------------------------------------------------------------------------
// Kernel 1: fused QKV projection, C = hs @ W + b. f16 tiles staged with
// global_load_async_to_lds_b128 (double buffered), fp32 WMMA accumulation.
// grid = (M/128, H/64, 3), block = 256 (8 waves, each owns a 16x64 strip)
// ---------------------------------------------------------------------------
__global__ __launch_bounds__(256) void qkv_gemm_f16wmma(
    const _Float16* __restrict__ hsf, const _Float16* __restrict__ Wt,
    const float* __restrict__ bq, const float* __restrict__ bk,
    const float* __restrict__ bv,
    _Float16* __restrict__ outq, _Float16* __restrict__ outk,
    _Float16* __restrict__ outv)
{
    const int z = blockIdx.z;
    const float* bias;  _Float16* out;
    if (z == 0)      { bias = bq; out = outq; }
    else if (z == 1) { bias = bk; out = outk; }
    else             { bias = bv; out = outv; }
    const _Float16* wt = Wt + (size_t)z * H * H;   // [n][k] layout

    __shared__ __attribute__((aligned(16))) _Float16 sA[2][128][40];  // [m][k]
    __shared__ __attribute__((aligned(16))) _Float16 sBt[2][64][40];  // [n][k]

    const int tid  = threadIdx.x;
    const int lane = tid & 31;
    const int wave = tid >> 5;
    const int rm0  = blockIdx.x * 128;
    const int n0   = blockIdx.y * 64;

    v8f acc[4];
#pragma unroll
    for (int t = 0; t < 4; ++t)
#pragma unroll
        for (int r = 0; r < 8; ++r) acc[t][r] = 0.0f;

    // async stage one 128x32 A tile + 64x32 B tile (3 issues per wave)
    auto issue_tile = [&](int buf, int k0) {
#pragma unroll
        for (int it = 0; it < 2; ++it) {            // 512 x 16B for A
            int tx = tid + it * 256;
            int r = tx >> 2, c = (tx & 3) << 3;
            unsigned lds = lds_addr32(&sA[buf][r][c]);
            unsigned long long ga = (unsigned long long)(uintptr_t)
                (hsf + (size_t)(rm0 + r) * H + k0 + c);
            asm volatile("global_load_async_to_lds_b128 %0, %1, off"
                         :: "v"(lds), "v"(ga) : "memory");
        }
        {                                           // 256 x 16B for B
            int n = tid >> 2, c = (tid & 3) << 3;
            unsigned lds = lds_addr32(&sBt[buf][n][c]);
            unsigned long long ga = (unsigned long long)(uintptr_t)
                (wt + (size_t)(n0 + n) * H + k0 + c);
            asm volatile("global_load_async_to_lds_b128 %0, %1, off"
                         :: "v"(lds), "v"(ga) : "memory");
        }
    };

    issue_tile(0, 0);

    for (int i = 0; i < H / 32; ++i) {
        const int buf = i & 1;
        const bool more = (i + 1) < (H / 32);
        if (more) issue_tile(buf ^ 1, (i + 1) * 32);
        if (more) asm volatile("s_wait_asynccnt 3" ::: "memory");
        else      asm volatile("s_wait_asynccnt 0" ::: "memory");
        __syncthreads();   // everyone's current-tile transfers landed

        // A fragment: 16x32 f16 (lane -> M, K contiguous at kb..kb+15)
        const int kb = (lane < 16) ? 0 : 16;
        const _Float16* Arow = &sA[buf][wave * 16 + (lane & 15)][0];
        v16h a;
#pragma unroll
        for (int q = 0; q < 4; ++q) {
            a[2 * q]     = Arow[kb + 2 * q];
            a[2 * q + 1] = Arow[kb + 2 * q + 1];
            a[8 + 2 * q] = Arow[8 + kb + 2 * q];
            a[9 + 2 * q] = Arow[8 + kb + 2 * q + 1];
        }
        const int kb2 = (lane < 16) ? 0 : 16;
#pragma unroll
        for (int t = 0; t < 4; ++t) {
            const _Float16* Bcol = &sBt[buf][t * 16 + (lane & 15)][0];
            v16h b;
#pragma unroll
            for (int j = 0; j < 8; ++j) {
                b[2 * j]     = Bcol[kb2 + 2 * j];
                b[2 * j + 1] = Bcol[kb2 + 2 * j + 1];
            }
            acc[t] = __builtin_amdgcn_wmma_f32_16x16x32_f16(
                false, a, false, b, (short)0, acc[t], false, false);
        }
        __syncthreads();   // buffer free for overwrite at step i+2
    }

    // epilogue: + bias, store f16
#pragma unroll
    for (int t = 0; t < 4; ++t) {
        int n = n0 + t * 16 + (lane & 15);
        float bval = bias[n];
#pragma unroll
        for (int r = 0; r < 8; ++r) {
            int m = wave * 16 + r + ((lane < 16) ? 0 : 8);
            out[(size_t)(rm0 + m) * H + n] = (_Float16)(acc[t][r] + bval);
        }
    }
}

// ---------------------------------------------------------------------------
// Kernel 2: local windowed attention with look-around, flash-style online
// softmax over 6 chunks of 64 keys. grid = (SEQ/WS, BS*NH), block = 256.
// Q and K staged with async-to-LDS; V staged with zero-fill transpose.
// ---------------------------------------------------------------------------
__global__ __launch_bounds__(256) void local_attn_wmma(
    const _Float16* __restrict__ qf, const _Float16* __restrict__ kf,
    const _Float16* __restrict__ vf, const unsigned char* __restrict__ kpm,
    float* __restrict__ out)
{
    const int w    = blockIdx.x;          // window index
    const int bh   = blockIdx.y;          // b*NH + h
    const int b    = bh / NH;
    const int h    = bh % NH;
    const int tid  = threadIdx.x;
    const int lane = tid & 31;
    const int wave = tid >> 5;

    __shared__ __attribute__((aligned(16))) _Float16 sQ[128][72];   // [row][hd]
    __shared__ __attribute__((aligned(16))) _Float16 sK[64][72];    // [key][hd]
    __shared__ __attribute__((aligned(16))) _Float16 sVt[64][72];   // [hd][key]
    __shared__ __attribute__((aligned(16))) _Float16 sP[8][16][72]; // wave P scratch
    __shared__ unsigned char sQm[128];
    __shared__ unsigned char sKm[384];

    // async-stage the whole Q window (4 issues per wave, 16B per lane)
#pragma unroll
    for (int it = 0; it < 4; ++it) {
        int tx = tid + it * 256;                 // 1024 x 16B
        int r = tx >> 3, c = (tx & 7) << 3;
        unsigned lds = lds_addr32(&sQ[r][c]);
        unsigned long long ga = (unsigned long long)(uintptr_t)
            (qf + ((size_t)(w * WS + r) * BS + b) * H + h * HD + c);
        asm volatile("global_load_async_to_lds_b128 %0, %1, off"
                     :: "v"(lds), "v"(ga) : "memory");
    }
    for (int e = tid; e < 128; e += 256)
        sQm[e] = kpm[(size_t)b * SEQ + w * WS + e];
    for (int e = tid; e < 384; e += 256) {
        int sk = (w - 1) * WS + e;
        sKm[e] = (sk < 0 || sk >= SEQ) ? 1 : kpm[(size_t)b * SEQ + sk];
    }

    const float scale = 0.125f;           // HD^-0.5
    const float NEG   = -3.0e38f;
    const int   halfoff = (lane < 16) ? 0 : 8;

    float m_i[8], l_i[8];
    v8f o[4];
#pragma unroll
    for (int r = 0; r < 8; ++r) { m_i[r] = NEG; l_i[r] = 0.0f; }
#pragma unroll
    for (int t = 0; t < 4; ++t)
#pragma unroll
        for (int r = 0; r < 8; ++r) o[t][r] = 0.0f;

    for (int c = 0; c < 6; ++c) {
        __syncthreads();                  // previous chunk fully consumed
        int s0 = (w - 1) * WS + c * 64;

        // K chunk via async copy; OOB rows not issued (mask kills them via
        // select, so stale LDS — even NaN — cannot propagate)
#pragma unroll
        for (int it = 0; it < 2; ++it) {
            int tx = tid + it * 256;             // 512 x 16B
            int kk = tx >> 3, cc = (tx & 7) << 3;
            int sk = s0 + kk;
            if (sk >= 0 && sk < SEQ) {
                unsigned lds = lds_addr32(&sK[kk][cc]);
                unsigned long long ga = (unsigned long long)(uintptr_t)
                    (kf + ((size_t)sk * BS + b) * H + h * HD + cc);
                asm volatile("global_load_async_to_lds_b128 %0, %1, off"
                             :: "v"(lds), "v"(ga) : "memory");
            }
        }
        // V chunk transposed with explicit zero-fill (p==0 keys need V==0)
        for (int e = tid; e < 64 * 64; e += 256) {
            int kk = e >> 6, d = e & 63;
            int sk = s0 + kk;
            _Float16 vvv = (_Float16)0.0f;
            if (sk >= 0 && sk < SEQ)
                vvv = vf[((size_t)sk * BS + b) * H + h * HD + d];
            sVt[d][kk] = vvv;
        }

        asm volatile("s_wait_asynccnt 0" ::: "memory");  // Q (first pass) + K
        __syncthreads();

        // ---- S = Q * K^T (16x64 per wave) ----
        v8f sacc[4];
#pragma unroll
        for (int t = 0; t < 4; ++t)
#pragma unroll
            for (int r = 0; r < 8; ++r) sacc[t][r] = 0.0f;

#pragma unroll
        for (int ks = 0; ks < 64; ks += 32) {
            const int kb = (lane < 16) ? 0 : 16;
            const _Float16* Arow = &sQ[wave * 16 + (lane & 15)][ks];
            v16h a;
#pragma unroll
            for (int q = 0; q < 4; ++q) {
                a[2 * q]     = Arow[kb + 2 * q];
                a[2 * q + 1] = Arow[kb + 2 * q + 1];
                a[8 + 2 * q] = Arow[8 + kb + 2 * q];
                a[9 + 2 * q] = Arow[8 + kb + 2 * q + 1];
            }
            const int kb2 = (lane < 16) ? 0 : 16;
#pragma unroll
            for (int t = 0; t < 4; ++t) {
                const _Float16* Brow = &sK[t * 16 + (lane & 15)][ks];
                v16h bb;
#pragma unroll
                for (int j = 0; j < 8; ++j) {
                    bb[2 * j]     = Brow[kb2 + 2 * j];
                    bb[2 * j + 1] = Brow[kb2 + 2 * j + 1];
                }
                sacc[t] = __builtin_amdgcn_wmma_f32_16x16x32_f16(
                    false, a, false, bb, (short)0, sacc[t], false, false);
            }
        }

        // ---- scale + mask + online softmax update ----
        float p[4][8];
#pragma unroll
        for (int r = 0; r < 8; ++r) {
            int  qrow = wave * 16 + r + halfoff;
            bool qm   = (sQm[qrow] != 0);
            float rowmax = NEG;
#pragma unroll
            for (int t = 0; t < 4; ++t) {
                int  kj  = c * 64 + t * 16 + (lane & 15);
                bool msk = qm || (sKm[kj] != 0);
                float sv = msk ? NEG : sacc[t][r] * scale;
                p[t][r]  = sv;
                rowmax   = fmaxf(rowmax, sv);
            }
#pragma unroll
            for (int off = 1; off < 16; off <<= 1)
                rowmax = fmaxf(rowmax, __shfl_xor(rowmax, off, 32));

            float mnew  = fmaxf(m_i[r], rowmax);
            float alpha = __expf(m_i[r] - mnew);   // finite sentinel -> no NaN
            float psum  = 0.0f;
#pragma unroll
            for (int t = 0; t < 4; ++t) {
                float e_ = __expf(p[t][r] - mnew);
                p[t][r] = e_;
                psum += e_;
            }
#pragma unroll
            for (int off = 1; off < 16; off <<= 1)
                psum += __shfl_xor(psum, off, 32);

            l_i[r] = l_i[r] * alpha + psum;
            m_i[r] = mnew;
#pragma unroll
            for (int t = 0; t < 4; ++t) o[t][r] *= alpha;
        }

        // ---- D-layout -> A-layout transpose of P through wave-private LDS ----
#pragma unroll
        for (int t = 0; t < 4; ++t)
#pragma unroll
            for (int r = 0; r < 8; ++r)
                sP[wave][r + halfoff][t * 16 + (lane & 15)] = (_Float16)p[t][r];

        asm volatile("s_wait_dscnt 0" ::: "memory");

        // ---- O += P * V ----
#pragma unroll
        for (int ks = 0; ks < 64; ks += 32) {
            const int kb = (lane < 16) ? 0 : 16;
            const _Float16* Prow = &sP[wave][lane & 15][ks];
            v16h a;
#pragma unroll
            for (int q = 0; q < 4; ++q) {
                a[2 * q]     = Prow[kb + 2 * q];
                a[2 * q + 1] = Prow[kb + 2 * q + 1];
                a[8 + 2 * q] = Prow[8 + kb + 2 * q];
                a[9 + 2 * q] = Prow[8 + kb + 2 * q + 1];
            }
            const int kb2 = (lane < 16) ? 0 : 16;
#pragma unroll
            for (int t = 0; t < 4; ++t) {
                const _Float16* Vrow = &sVt[t * 16 + (lane & 15)][0]; // [hd][key]
                v16h bb;
#pragma unroll
                for (int j = 0; j < 8; ++j) {
                    bb[2 * j]     = Vrow[ks + kb2 + 2 * j];
                    bb[2 * j + 1] = Vrow[ks + kb2 + 2 * j + 1];
                }
                o[t] = __builtin_amdgcn_wmma_f32_16x16x32_f16(
                    false, a, false, bb, (short)0, o[t], false, false);
            }
        }
    }

    // ---- normalize and store (seq, bs, H) fp32 ----
#pragma unroll
    for (int t = 0; t < 4; ++t) {
        int d = t * 16 + (lane & 15);
#pragma unroll
        for (int r = 0; r < 8; ++r) {
            int m = wave * 16 + r + halfoff;
            int s = w * WS + m;
            out[((size_t)s * BS + b) * H + h * HD + d] = o[t][r] / l_i[r];
        }
    }
}

// ---------------------------------------------------------------------------
// Kernel 3: global token (position 0) attention over all SEQ keys.
// grid = BS*NH, block = 256. Overwrites out row s==0 (runs after kernel 2).
// ---------------------------------------------------------------------------
__global__ __launch_bounds__(256) void global_attn_kernel(
    const float* __restrict__ hs, const float* __restrict__ Wgq,
    const float* __restrict__ bgq,
    const _Float16* __restrict__ kf, const _Float16* __restrict__ vf,
    const unsigned char* __restrict__ kpm, float* __restrict__ out)
{
    const int bh = blockIdx.x;
    const int b  = bh / NH;
    const int h  = bh % NH;
    const int tid = threadIdx.x;

    __shared__ float sgq[64];
    __shared__ float sc[SEQ];
    __shared__ float sred[256];
    __shared__ float spart[4][64];

    if (tid < 64) sgq[tid] = 0.0f;
    __syncthreads();
    {   // gq = hs[s=0,b,:] @ Wgq[:, h*64+d] (fp32, split over 4 groups)
        int d = tid & 63, g = tid >> 6;
        float acc = 0.0f;
        for (int j = g * 192; j < (g + 1) * 192; ++j)
            acc += hs[(size_t)b * H + j] * Wgq[(size_t)j * H + h * HD + d];
        atomicAdd(&sgq[d], acc);
    }
    __syncthreads();
    if (tid < 64) sgq[tid] += bgq[h * HD + tid];
    __syncthreads();

    // scores over all keys
    float lmax = -3.0e38f;
    for (int s = tid; s < SEQ; s += 256) {
        size_t base = ((size_t)s * BS + b) * H + h * HD;
        float dot = 0.0f;
        for (int d = 0; d < HD; ++d) dot += sgq[d] * (float)kf[base + d];
        float v_ = (kpm[(size_t)b * SEQ + s] != 0) ? -3.0e38f : dot * 0.125f;
        sc[s] = v_;
        lmax = fmaxf(lmax, v_);
    }
    sred[tid] = lmax;
    __syncthreads();
    for (int st = 128; st > 0; st >>= 1) {
        if (tid < st) sred[tid] = fmaxf(sred[tid], sred[tid + st]);
        __syncthreads();
    }
    float gmax = sred[0];
    __syncthreads();

    float lsum = 0.0f;
    for (int s = tid; s < SEQ; s += 256) {
        float e_ = __expf(sc[s] - gmax);
        sc[s] = e_;
        lsum += e_;
    }
    sred[tid] = lsum;
    __syncthreads();
    for (int st = 128; st > 0; st >>= 1) {
        if (tid < st) sred[tid] += sred[tid + st];
        __syncthreads();
    }
    float gsum = sred[0];
    __syncthreads();

    {   // out[d] = sum_s p[s] * v[s][d] / gsum
        int d = tid & 63, g = tid >> 6;
        float acc = 0.0f;
        for (int s = g; s < SEQ; s += 4)
            acc += sc[s] * (float)vf[((size_t)s * BS + b) * H + h * HD + d];
        spart[g][d] = acc;
    }
    __syncthreads();
    if (tid < 64) {
        float tot = spart[0][tid] + spart[1][tid] + spart[2][tid] + spart[3][tid];
        out[(size_t)b * H + h * HD + tid] = tot / gsum;   // row s == 0
    }
}

// ---------------------------------------------------------------------------
extern "C" void kernel_launch(void* const* d_in, const int* in_sizes, int n_in,
                              void* d_out, int out_size, void* d_ws, size_t ws_size,
                              hipStream_t stream) {
    (void)in_sizes; (void)n_in; (void)out_size; (void)ws_size;
    const float*         hs  = (const float*)d_in[0];
    const unsigned char* kpm = (const unsigned char*)d_in[1];
    const float* Wq  = (const float*)d_in[2];
    const float* bq  = (const float*)d_in[3];
    const float* Wk  = (const float*)d_in[4];
    const float* bk  = (const float*)d_in[5];
    const float* Wv  = (const float*)d_in[6];
    const float* bv  = (const float*)d_in[7];
    const float* Wgq = (const float*)d_in[8];
    const float* bgq = (const float*)d_in[9];
    float* out = (float*)d_out;

    _Float16* qf  = (_Float16*)d_ws;                 // 16384 x 768 f16 each
    _Float16* kf  = qf  + (size_t)MTOT * H;
    _Float16* vf  = kf  + (size_t)MTOT * H;
    _Float16* hsf = vf  + (size_t)MTOT * H;          // f16 activations
    _Float16* Wt  = hsf + (size_t)MTOT * H;          // 3 x [n][k] f16 weights

    convert_hs<<<2048, 256, 0, stream>>>(hs, hsf);
    convert_w<<<1024, 256, 0, stream>>>(Wq, Wk, Wv, Wt);
    qkv_gemm_f16wmma<<<dim3(MTOT / 128, H / 64, 3), 256, 0, stream>>>(
        hsf, Wt, bq, bk, bv, qf, kf, vf);
    local_attn_wmma<<<dim3(SEQ / WS, BS * NH), 256, 0, stream>>>(
        qf, kf, vf, kpm, out);
    global_attn_kernel<<<BS * NH, 256, 0, stream>>>(
        hs, Wgq, bgq, kf, vf, kpm, out);
}
